// SimpleMHAJAX_50629074485296
// MI455X (gfx1250) — matrix operand order repaired and checked by reading, not compiled, not run on hardware
//
#include <hip/hip_runtime.h>

// ---------------------------------------------------------------------------
// Causal MHA forward for MI455X (gfx1250), bf16 WMMA (v_wmma_f32_16x16x32_bf16)
//   1) convert x -> bf16; transpose-convert Wqkv, Wout -> bf16 [N][K]
//   2) WMMA GEMM (128x128 tile, 8 waves, double-buffered LDS): QKV projection
//   3) flash attention: S^T = K Q^T so softmax reduces in-lane (2 shfls/block)
//   4) WMMA GEMM: out = AO @ WoutT -> f32 d_out
// ---------------------------------------------------------------------------

typedef __bf16 bf16_t;
typedef __attribute__((ext_vector_type(16))) __bf16 v16bf;
typedef __attribute__((ext_vector_type(8)))  __bf16 bf16x8;
typedef __attribute__((ext_vector_type(4)))  __bf16 bf16x4;
typedef __attribute__((ext_vector_type(8)))  float  v8f;

#define SEQ    4096
#define DMODEL 1024
#define HEADS  16
#define HD     64

static __device__ __forceinline__ v8f wmma_bf16(v16bf a, v16bf b, v8f c) {
  return __builtin_amdgcn_wmma_f32_16x16x32_bf16(false, a, false, b,
                                                 (short)0, c, false, false);
}

static __device__ __forceinline__ v16bf join8(bf16x8 lo, bf16x8 hi) {
  return __builtin_shufflevector(lo, hi, 0,1,2,3,4,5,6,7,8,9,10,11,12,13,14,15);
}

// ---------------------------------------------------------------------------
__global__ void cvt_f32_to_bf16_v4(const float4* __restrict__ src,
                                   bf16x4* __restrict__ dst, int n4) {
  int i = blockIdx.x * blockDim.x + threadIdx.x;
  if (i < n4) {
    float4 v = src[i];
    bf16x4 o;
    o[0] = (bf16_t)v.x; o[1] = (bf16_t)v.y; o[2] = (bf16_t)v.z; o[3] = (bf16_t)v.w;
    dst[i] = o;
  }
}

// transpose-convert: src[K][N] f32 -> dst[N][K] bf16
__global__ void cvtT_f32_to_bf16(const float* __restrict__ src,
                                 bf16_t* __restrict__ dst, int K, int N) {
  int n = blockIdx.x * blockDim.x + threadIdx.x;
  int k = blockIdx.y * blockDim.y + threadIdx.y;
  if (n < N && k < K) dst[(size_t)n * K + k] = (bf16_t)src[(size_t)k * N + n];
}

// ---------------------------------------------------------------------------
// WMMA GEMM: C[M][N] = A[M][Kd] * B^T  (B stored [N][Kd], both bf16)
// 256 threads = 8 waves; 128x128 block tile; wave = 32x64 (2x4 WMMA tiles).
// Double-buffered LDS. MODE 0: QKV scatter epilogue. MODE 1: f32 -> Cout.
// ---------------------------------------------------------------------------
template <int M, int N, int Kd, int MODE>
__global__ __launch_bounds__(256) void gemm_wmma_bf16(
    const bf16_t* __restrict__ A, const bf16_t* __restrict__ B,
    bf16_t* __restrict__ Qb, bf16_t* __restrict__ Kb, bf16_t* __restrict__ Vt,
    float* __restrict__ Cout)
{
  __shared__ __align__(16) bf16_t As[2][128][40];
  __shared__ __align__(16) bf16_t Bs[2][128][40];

  const int tid  = threadIdx.x;
  const int lane = tid & 31;
  const int wave = tid >> 5;
  const int lh   = lane >> 4;
  const int ll   = lane & 15;
  const int m0   = blockIdx.y * 128;
  const int n0   = blockIdx.x * 128;
  const int wm   = (wave >> 1) * 32;
  const int wn   = (wave & 1) * 64;

  v8f acc[2][4];
  #pragma unroll
  for (int i = 0; i < 2; ++i)
    #pragma unroll
    for (int j = 0; j < 4; ++j)
      #pragma unroll
      for (int e = 0; e < 8; ++e) acc[i][j][e] = 0.0f;

  const int lr = tid >> 2;             // 0..63
  const int lc = (tid & 3) * 8;        // 0,8,16,24

  const bf16_t* Ag = &A[(size_t)(m0 + lr) * Kd + lc];
  const bf16_t* Bg = &B[(size_t)(n0 + lr) * Kd + lc];

  bf16x8 ar0, ar1, br0, br1;
  auto load_regs = [&](int k0) {
    ar0 = *reinterpret_cast<const bf16x8*>(Ag + k0);
    ar1 = *reinterpret_cast<const bf16x8*>(Ag + (size_t)64 * Kd + k0);
    br0 = *reinterpret_cast<const bf16x8*>(Bg + k0);
    br1 = *reinterpret_cast<const bf16x8*>(Bg + (size_t)64 * Kd + k0);
  };
  auto store_lds = [&](int buf) {
    *reinterpret_cast<bf16x8*>(&As[buf][lr][lc])      = ar0;
    *reinterpret_cast<bf16x8*>(&As[buf][lr + 64][lc]) = ar1;
    *reinterpret_cast<bf16x8*>(&Bs[buf][lr][lc])      = br0;
    *reinterpret_cast<bf16x8*>(&Bs[buf][lr + 64][lc]) = br1;
  };

  load_regs(0);
  store_lds(0);
  __syncthreads();

  int cur = 0;
  for (int k0 = 0; k0 < Kd; k0 += 32) {
    const bool more = (k0 + 32) < Kd;
    if (more) {
      load_regs(k0 + 32);                          // overlap with WMMA below
      if (k0 + 64 < Kd) {                          // global_prefetch_b8
        __builtin_prefetch(Ag + k0 + 64, 0, 1);
        __builtin_prefetch(Bg + k0 + 64, 0, 1);
      }
    }

    v16bf af[2], bfr[4];
    const int kb  = lh * 8;
    const int kb2 = lh * 16;
    #pragma unroll
    for (int i = 0; i < 2; ++i) {
      const bf16_t* row = &As[cur][wm + i * 16 + ll][0];
      af[i] = join8(*reinterpret_cast<const bf16x8*>(row + kb),
                    *reinterpret_cast<const bf16x8*>(row + kb + 16));
    }
    #pragma unroll
    for (int j = 0; j < 4; ++j) {
      const bf16_t* row = &Bs[cur][wn + j * 16 + ll][0];
      bfr[j] = join8(*reinterpret_cast<const bf16x8*>(row + kb),
                     *reinterpret_cast<const bf16x8*>(row + kb + 16));
    }
    #pragma unroll
    for (int i = 0; i < 2; ++i)
      #pragma unroll
      for (int j = 0; j < 4; ++j)
        acc[i][j] = wmma_bf16(af[i], bfr[j], acc[i][j]);

    if (more) {
      store_lds(cur ^ 1);
      __syncthreads();
      cur ^= 1;
    }
  }

  #pragma unroll
  for (int j = 0; j < 4; ++j) {
    const int ntile = n0 + wn + j * 16;
    #pragma unroll
    for (int i = 0; i < 2; ++i) {
      const int mbase = m0 + wm + i * 16 + 8 * lh;
      if (MODE == 0) {
        const int mat = ntile >> 10;        // 0=Q 1=K 2=V
        const int h   = (ntile & 1023) >> 6;
        const int d   = (ntile & 63) + ll;
        if (mat == 0) {
          bf16_t* p = &Qb[((size_t)h * SEQ + mbase) * HD + d];
          #pragma unroll
          for (int r = 0; r < 8; ++r) p[r * HD] = (bf16_t)acc[i][j][r];
        } else if (mat == 1) {
          bf16_t* p = &Kb[((size_t)h * SEQ + mbase) * HD + d];
          #pragma unroll
          for (int r = 0; r < 8; ++r) p[r * HD] = (bf16_t)acc[i][j][r];
        } else {                            // V transposed: [h][d][seq]
          bf16_t* p = &Vt[((size_t)h * HD + d) * SEQ + mbase];
          #pragma unroll
          for (int r = 0; r < 8; ++r) p[r] = (bf16_t)acc[i][j][r];
        }
      } else {
        float* p = &Cout[(size_t)mbase * N + ntile + ll];
        #pragma unroll
        for (int r = 0; r < 8; ++r) p[(size_t)r * N] = acc[i][j][r];
      }
    }
  }
}

// ---------------------------------------------------------------------------
// Causal flash attention. Block = (head, 64 q rows); wave = 16 q rows.
// S^T = K_tile @ Q^T: each lane owns one q column (q = q0 + lane%16), so the
// softmax max/sum reduce over registers + ONE shfl_xor(16) per block.
// All four K fragments of a 32-key block are preloaded before the S WMMAs so
// the loads clause together and the waits stage down instead of blocking.
// ---------------------------------------------------------------------------
__global__ __launch_bounds__(128) void attn_wmma_bf16(
    const bf16_t* __restrict__ Qb, const bf16_t* __restrict__ Kb,
    const bf16_t* __restrict__ Vt, bf16_t* __restrict__ AO)
{
  __shared__ __align__(16) bf16_t Ps[4][16][40];   // per-wave 16(q) x 32(key) P
  __shared__ __align__(16) float  Sc[4][16];       // per-wave rescale broadcast
  __shared__ __align__(16) float  Li[4][16];       // per-wave 1/l broadcast

  const int lane = threadIdx.x & 31;
  const int wave = threadIdx.x >> 5;
  const int lh   = lane >> 4;
  const int ll   = lane & 15;
  const int h    = blockIdx.y;
  const int q0   = blockIdx.x * 64 + wave * 16;
  const int kb   = lh * 8;

  // ---- Q as B fragments of S^T (lane = q column, d contiguous), loaded once
  const bf16_t* Qrow = Qb + ((size_t)h * SEQ + q0 + ll) * HD;
  v16bf qbf[2];
  #pragma unroll
  for (int half = 0; half < 2; ++half) {
    const int off = half * 32 + lh * 8;
    qbf[half] = join8(*reinterpret_cast<const bf16x8*>(Qrow + off),
                      *reinterpret_cast<const bf16x8*>(Qrow + off + 16));
  }

  v8f o[4];
  #pragma unroll
  for (int t = 0; t < 4; ++t)
    #pragma unroll
    for (int e = 0; e < 8; ++e) o[t][e] = 0.0f;
  float mst = -1.0e30f, lst = 0.0f;    // softmax state for q = q0 + ll

  const bf16_t* Kh = Kb + (size_t)h * SEQ * HD;
  const bf16_t* Vh = Vt + (size_t)h * HD * SEQ;
  const int q    = q0 + ll;
  const int kend = q0 + 16;            // exclusive key bound (causal)
  const float scale = 0.125f;          // 1/sqrt(64)

  for (int kb0 = 0; kb0 < kend; kb0 += 32) {
    if (kb0 + 32 < kend)                              // global_prefetch_b8
      __builtin_prefetch(Kh + (size_t)(kb0 + 32 + ll) * HD, 0, 1);

    const bool have2 = (kb0 + 16) < kend;

    // ---- preload ALL K fragments for this 32-key block (clauses together)
    const bf16_t* Krow0 = Kh + (size_t)(kb0 + ll) * HD;
    v16bf ka0[2], ka1[2];
    #pragma unroll
    for (int half = 0; half < 2; ++half) {
      const int off = half * 32 + kb;
      ka0[half] = join8(*reinterpret_cast<const bf16x8*>(Krow0 + off),
                        *reinterpret_cast<const bf16x8*>(Krow0 + off + 16));
    }
    if (have2) {
      const bf16_t* Krow1 = Kh + (size_t)(kb0 + 16 + ll) * HD;
      #pragma unroll
      for (int half = 0; half < 2; ++half) {
        const int off = half * 32 + kb;
        ka1[half] = join8(*reinterpret_cast<const bf16x8*>(Krow1 + off),
                          *reinterpret_cast<const bf16x8*>(Krow1 + off + 16));
      }
    }

    // ---- S^T tiles: A = K rows (16 keys x 32 d), B = Q^T
    v8f t0, t1;
    {
      v8f s;
      #pragma unroll
      for (int e = 0; e < 8; ++e) s[e] = 0.0f;
      s = wmma_bf16(ka0[0], qbf[0], s);
      s = wmma_bf16(ka0[1], qbf[1], s);
      t0 = s;
    }
    if (have2) {
      v8f s;
      #pragma unroll
      for (int e = 0; e < 8; ++e) s[e] = 0.0f;
      s = wmma_bf16(ka1[0], qbf[0], s);
      s = wmma_bf16(ka1[1], qbf[1], s);
      t1 = s;
    }

    // ---- per-lane scores: x[i] = score(key = kb0 + i + 8*lh (+16), q)
    float x[16];
    #pragma unroll
    for (int r = 0; r < 8; ++r) {
      float v = t0[r] * scale;
      if (kb0 + r + 8 * lh > q) v = -1.0e30f;
      x[r] = v;
    }
    #pragma unroll
    for (int r = 0; r < 8; ++r) {
      float v = -1.0e30f;
      if (have2) {
        v = t1[r] * scale;
        if (kb0 + 16 + r + 8 * lh > q) v = -1.0e30f;
      }
      x[8 + r] = v;
    }

    // ---- online softmax: register reduce + one half-wave exchange
    float mx = x[0];
    #pragma unroll
    for (int i = 1; i < 16; ++i) mx = fmaxf(mx, x[i]);
    mx = fmaxf(mx, __shfl_xor(mx, 16, 32));
    const float mnew = fmaxf(mst, mx);
    const float scf  = __expf(mst - mnew);
    mst = mnew;

    bf16x8 pk0, pk1;
    float ps = 0.0f;
    #pragma unroll
    for (int i = 0; i < 8; ++i) {
      const float p = __expf(x[i] - mnew);
      ps += p; pk0[i] = (bf16_t)p;
    }
    #pragma unroll
    for (int i = 0; i < 8; ++i) {
      const float p = __expf(x[8 + i] - mnew);
      ps += p; pk1[i] = (bf16_t)p;
    }
    ps += __shfl_xor(ps, 16, 32);
    lst = lst * scf + ps;

    // ---- P^T store: lane l -> row q=ll, keys (8*lh..8*lh+7) and +16
    *reinterpret_cast<bf16x8*>(&Ps[wave][ll][8 * lh])      = pk0;
    *reinterpret_cast<bf16x8*>(&Ps[wave][ll][16 + 8 * lh]) = pk1;

    // ---- broadcast rescale factor; O rows 8*lh..8*lh+7 are contiguous
    Sc[wave][ll] = scf;
    const float4 f0 = *reinterpret_cast<const float4*>(&Sc[wave][8 * lh]);
    const float4 f1 = *reinterpret_cast<const float4*>(&Sc[wave][8 * lh + 4]);
    const float sarr[8] = {f0.x, f0.y, f0.z, f0.w, f1.x, f1.y, f1.z, f1.w};
    #pragma unroll
    for (int t = 0; t < 4; ++t)
      #pragma unroll
      for (int r = 0; r < 8; ++r) o[t][r] *= sarr[r];

    // ---- P as A fragment (wave-private LDS, in-order within wave)
    const bf16_t* prow = &Ps[wave][ll][0];
    v16bf pa = join8(*reinterpret_cast<const bf16x8*>(prow + kb),
                     *reinterpret_cast<const bf16x8*>(prow + kb + 16));
    // ---- PV: 4 d-tiles, B fragments contiguous from transposed V
    #pragma unroll
    for (int t = 0; t < 4; ++t) {
      const bf16_t* vrow = Vh + (size_t)(t * 16 + ll) * SEQ + kb0 + kb;
      v16bf vf = join8(*reinterpret_cast<const bf16x8*>(vrow),
                       *reinterpret_cast<const bf16x8*>(vrow + 16));
      o[t] = wmma_bf16(pa, vf, o[t]);
    }
  }

  // ---- normalize (broadcast 1/l the same way) + store AO[SEQ][DMODEL] bf16
  Li[wave][ll] = 1.0f / lst;
  const float4 g0 = *reinterpret_cast<const float4*>(&Li[wave][8 * lh]);
  const float4 g1 = *reinterpret_cast<const float4*>(&Li[wave][8 * lh + 4]);
  const float iarr[8] = {g0.x, g0.y, g0.z, g0.w, g1.x, g1.y, g1.z, g1.w};
  #pragma unroll
  for (int r = 0; r < 8; ++r) {
    const int qrow = q0 + r + 8 * lh;
    #pragma unroll
    for (int t = 0; t < 4; ++t)
      AO[(size_t)qrow * DMODEL + h * HD + t * 16 + ll] = (bf16_t)(o[t][r] * iarr[r]);
  }
}

// ---------------------------------------------------------------------------
extern "C" void kernel_launch(void* const* d_in, const int* in_sizes, int n_in,
                              void* d_out, int out_size, void* d_ws, size_t ws_size,
                              hipStream_t stream) {
  const float* x    = (const float*)d_in[0];   // [4096,1024]
  const float* Wqkv = (const float*)d_in[1];   // [1024,3072]
  const float* Wout = (const float*)d_in[2];   // [1024,1024]
  float* out = (float*)d_out;                  // [4096,1024]

  char* ws = (char*)d_ws;
  bf16_t* Xb    = (bf16_t*)ws;                 ws += (size_t)SEQ * DMODEL * 2;
  bf16_t* WqkvT = (bf16_t*)ws;                 ws += (size_t)3 * DMODEL * DMODEL * 2;
  bf16_t* WoutT = (bf16_t*)ws;                 ws += (size_t)DMODEL * DMODEL * 2;
  bf16_t* Qb    = (bf16_t*)ws;                 ws += (size_t)SEQ * DMODEL * 2;
  bf16_t* Kb    = (bf16_t*)ws;                 ws += (size_t)SEQ * DMODEL * 2;
  bf16_t* Vt    = (bf16_t*)ws;                 ws += (size_t)SEQ * DMODEL * 2;
  bf16_t* AO    = (bf16_t*)ws;                 ws += (size_t)SEQ * DMODEL * 2;

  {
    int n4 = SEQ * DMODEL / 4;
    cvt_f32_to_bf16_v4<<<(n4 + 255) / 256, 256, 0, stream>>>(
        (const float4*)x, (bf16x4*)Xb, n4);
    cvtT_f32_to_bf16<<<dim3(3 * DMODEL / 32, DMODEL / 8), dim3(32, 8), 0, stream>>>(
        Wqkv, WqkvT, DMODEL, 3 * DMODEL);
    cvtT_f32_to_bf16<<<dim3(DMODEL / 32, DMODEL / 8), dim3(32, 8), 0, stream>>>(
        Wout, WoutT, DMODEL, DMODEL);
  }
  gemm_wmma_bf16<SEQ, 3 * DMODEL, DMODEL, 0>
      <<<dim3(3 * DMODEL / 128, SEQ / 128), 256, 0, stream>>>(
          Xb, WqkvT, Qb, Kb, Vt, nullptr);
  attn_wmma_bf16<<<dim3(SEQ / 64, HEADS), 128, 0, stream>>>(Qb, Kb, Vt, AO);
  gemm_wmma_bf16<SEQ, DMODEL, DMODEL, 1>
      <<<dim3(DMODEL / 128, SEQ / 128), 256, 0, stream>>>(
          AO, WoutT, nullptr, nullptr, nullptr, out);
}
